// HiddenMarkovModel_26585847562493
// MI455X (gfx1250) — matrix operand order repaired and checked by reading, not compiled, not run on hardware
//
#include <hip/hip_runtime.h>
#include <math.h>

// ---------------------------------------------------------------------------
// HMM scaled-forward log-likelihood on gfx1250 (MI455X), WMMA-centric.
//
// Phases:
//  1) k_prep    : A = softmax(WA) rows, zero-padded to 112x112 f32.
//  2) k_rowsum  : rowsum[k] = sum_v exp(ThetaB[k] . E[v])   (100x100000x256 GEMM)
//                 bf16 hi/lo split WMMA, E streamed once (HBM-bound, ~102 MB).
//  3) k_emit    : s~[t][k] = normalized emit*tau step vectors + log sigma_t.
//  4) k_group   : P~_g = prod_{t in group} (A diag(s~_t)) for 400 groups of 50,
//                 chained 112x112 WMMA products; diag = per-lane column scale.
//  5) k_combine : serial alpha <- alpha P~_g over 400 groups + final EOS step,
//                 sums all accumulated logs -> scalar output.
// ---------------------------------------------------------------------------

typedef __attribute__((ext_vector_type(16))) __bf16 v16bf;
typedef __attribute__((ext_vector_type(8)))  float  v8f;

#define KTAG    100
#define KP      112      // padded tag dim (7 tiles of 16)
#define BOS_TAG 99
#define EOS_TAG 98
#define DIMD    256
#define TLEN    20000
#define NTTILE  1250     // TLEN/16
#define NVTILE  6250     // VOCAB/16
#define VTPB    10       // v-tiles per block in k_rowsum
#define NVBLK   625      // NVTILE/VTPB
#define GROUP_L 50
#define NGROUP  400      // TLEN/GROUP_L
#define PSTRIDE 130      // LDS row stride (bf16): 65 dwords -> conflict-friendly

__device__ __forceinline__ v8f wmma_bf(v16bf a, v16bf b, v8f c) {
  return __builtin_amdgcn_wmma_f32_16x16x32_bf16(false, a, false, b, (short)0, c,
                                                 false, false);
}
// A fragment (16x32 MxK): element e of lane (half = lane>>4) holds K =
// (e&7) + 16*(e>>3) + 8*half   (ISA 7.12.2, 16-bit A layout)
__device__ __forceinline__ int a_kidx(int e, int half) {
  return (e & 7) + ((e >> 3) << 4) + (half << 3);
}
// B fragment (32x16 KxN): element e holds K = (e&7) + 16*(e>>3) + 8*half (matches A)
__device__ __forceinline__ int b_kidx(int e, int half) { return (e & 7) + ((e >> 3) << 4) + (half << 3); }

__device__ __forceinline__ void splitf(float x, __bf16 &hi, __bf16 &lo) {
  hi = (__bf16)x;
  lo = (__bf16)(x - (float)hi);
}

// ---------------------------------------------------------------- k_prep ----
__global__ void k_prep(const float* __restrict__ WA, float* __restrict__ Asoft) {
  int row = threadIdx.x;
  if (row >= KP) return;
  if (row >= KTAG) {
    for (int j = 0; j < KP; ++j) Asoft[row * KP + j] = 0.f;
    return;
  }
  float m = -INFINITY;
  for (int j = 0; j < KTAG; ++j) m = fmaxf(m, WA[row * KTAG + j]);
  float s = 0.f;
  for (int j = 0; j < KTAG; ++j) s += expf(WA[row * KTAG + j] - m);
  float inv = 1.f / s;
  for (int j = 0; j < KP; ++j) {
    float v = (j < KTAG) ? expf(WA[row * KTAG + j] - m) * inv : 0.f;
    Asoft[row * KP + j] = v;   // column BOS_TAG: exp(-inf)=0 automatically
  }
}

// -------------------------------------------------------------- k_rowsum ----
// Block: 7 waves (224 thr); wave c owns tag tile c; block covers VTPB v-tiles.
// D[v_tile(16) x tag_tile(16)] = E_rows x ThetaB^T, K-dim = 256 (8 WMMA chunks).
__global__ void __launch_bounds__(224)
k_rowsum(const float* __restrict__ ThetaB, const float* __restrict__ E,
         float* __restrict__ rowsum) {
  __shared__ float rsLoc[KP];
  int tid = threadIdx.x;
  if (tid < KP) rsLoc[tid] = 0.f;
  __syncthreads();
  int lane = tid & 31, c = tid >> 5;
  int half = lane >> 4, l15 = lane & 15;
  int tag = 16 * c + l15;

  // B-operand: ThetaB^T hi/lo fragments, resident for whole kernel (128 VGPRs)
  v16bf Bh[8], Bl[8];
#pragma unroll
  for (int kk = 0; kk < 8; ++kk) {
#pragma unroll
    for (int e = 0; e < 16; ++e) {
      int d = 32 * kk + b_kidx(e, half);
      float x = (tag < KTAG) ? ThetaB[tag * DIMD + d] : 0.f;
      __bf16 hi, lo; splitf(x, hi, lo);
      Bh[kk][e] = hi; Bl[kk][e] = lo;
    }
  }

  float partial = 0.f;
  int vt0 = blockIdx.x * VTPB;
  for (int it = 0; it < VTPB; ++it) {
    size_t vrow = (size_t)((vt0 + it) * 16 + l15);
    v8f acc = {0.f, 0.f, 0.f, 0.f, 0.f, 0.f, 0.f, 0.f};
#pragma unroll
    for (int kk = 0; kk < 8; ++kk) {
      v16bf ah, al;
#pragma unroll
      for (int e = 0; e < 16; ++e) {
        float x = E[vrow * DIMD + 32 * kk + a_kidx(e, half)];
        __bf16 hi, lo; splitf(x, hi, lo);
        ah[e] = hi; al[e] = lo;
      }
      acc = wmma_bf(ah, Bh[kk], acc);   // hi*hi
      acc = wmma_bf(ah, Bl[kk], acc);   // hi*lo
      acc = wmma_bf(al, Bh[kk], acc);   // lo*hi
    }
#pragma unroll
    for (int v = 0; v < 8; ++v) partial += expf(acc[v]);
  }
  atomicAdd(&rsLoc[tag], partial);
  __syncthreads();
  if (tid < KP) atomicAdd(&rowsum[tid], rsLoc[tid]);
}

// ---------------------------------------------------------------- k_emit ----
// One block per t-tile (16 positions); wave c owns tag tile c.
// Produces normalized step vectors s~[t][0..111] and log sigma_t.
__global__ void __launch_bounds__(224)
k_emit(const float* __restrict__ ThetaB, const float* __restrict__ E,
       const int* __restrict__ words, const int* __restrict__ tags,
       const float* __restrict__ rowsum, float* __restrict__ S,
       float* __restrict__ logsig) {
  __shared__ float sloc[16][KP];
  __shared__ float siginv[16];
  int tid = threadIdx.x;
  int lane = tid & 31, c = tid >> 5;
  int half = lane >> 4, l15 = lane & 15;
  int tag = 16 * c + l15;
  int tbase = blockIdx.x * 16;

  v16bf Bh[8], Bl[8];
#pragma unroll
  for (int kk = 0; kk < 8; ++kk) {
#pragma unroll
    for (int e = 0; e < 16; ++e) {
      int d = 32 * kk + b_kidx(e, half);
      float x = (tag < KTAG) ? ThetaB[tag * DIMD + d] : 0.f;
      __bf16 hi, lo; splitf(x, hi, lo);
      Bh[kk][e] = hi; Bl[kk][e] = lo;
    }
  }

  size_t wrow = (size_t)words[tbase + l15];   // gathered embedding row for M=l15
  v8f acc = {0.f, 0.f, 0.f, 0.f, 0.f, 0.f, 0.f, 0.f};
#pragma unroll
  for (int kk = 0; kk < 8; ++kk) {
    v16bf ah, al;
#pragma unroll
    for (int e = 0; e < 16; ++e) {
      float x = E[wrow * DIMD + 32 * kk + a_kidx(e, half)];
      __bf16 hi, lo; splitf(x, hi, lo);
      ah[e] = hi; al[e] = lo;
    }
    acc = wmma_bf(ah, Bh[kk], acc);
    acc = wmma_bf(ah, Bl[kk], acc);
    acc = wmma_bf(al, Bh[kk], acc);
  }

  float invr = 1.f / rowsum[tag];
#pragma unroll
  for (int v = 0; v < 8; ++v) {
    int r = v + 8 * half;                 // C-frag row
    int tg = tags[tbase + r];
    // emit zeroed for EOS/BOS rows & pad; tau one-hot for supervised positions
    bool on = (tag < EOS_TAG) && ((tg >= KTAG) || (tag == (tg % (KTAG - 2))));
    sloc[r][tag] = on ? expf(acc[v]) * invr : 0.f;
  }
  __syncthreads();
  if (tid < 16) {
    float s = 0.f;
    for (int k2 = 0; k2 < KP; ++k2) s += sloc[tid][k2];
    logsig[tbase + tid] = logf(s);
    siginv[tid] = 1.f / s;
  }
  __syncthreads();
  for (int idx = tid; idx < 16 * KP; idx += 224) {
    int r = idx / KP, k2 = idx % KP;
    S[(size_t)(tbase + r) * KP + k2] = sloc[r][k2] * siginv[r];
  }
}

// --------------------------------------------------------------- k_group ----
// Block g: P~ = prod_{j=0..49} (A diag(s~_{g*50+j})) with per-step mass renorm.
// P kept in LDS as bf16 hi/lo (A-operand layout source); A_hmm B-fragments are
// loop-invariant registers; diag(s~) is a per-lane column scale on C-frags.
__global__ void __launch_bounds__(224)
k_group(const float* __restrict__ Asoft, const float* __restrict__ S,
        float* __restrict__ Pout, float* __restrict__ logG) {
  __shared__ __bf16 Phi[KP][PSTRIDE];
  __shared__ __bf16 Plo[KP][PSTRIDE];
  __shared__ float red[256];
  int tid = threadIdx.x;
  int lane = tid & 31, c = tid >> 5;
  int half = lane >> 4, l15 = lane & 15;
  int n = 16 * c + l15;                   // this lane's output column
  int g = blockIdx.x;

  for (int idx = tid; idx < KP * PSTRIDE; idx += 224) {
    (&Phi[0][0])[idx] = (__bf16)0.f;
    (&Plo[0][0])[idx] = (__bf16)0.f;
  }
  if (tid < 32) red[224 + tid] = 0.f;     // tree-reduce padding
  __syncthreads();
  if (tid < KP) Phi[tid][tid] = (__bf16)1.f;   // P = I (exact in bf16)
  __syncthreads();

  // Loop-invariant B-operand: A_hmm columns [16c..16c+15], K padded to 128.
  v16bf Bh[4], Bl[4];
#pragma unroll
  for (int kk = 0; kk < 4; ++kk) {
#pragma unroll
    for (int e = 0; e < 16; ++e) {
      int k2 = 32 * kk + b_kidx(e, half);
      float x = (k2 < KP) ? Asoft[k2 * KP + n] : 0.f;
      __bf16 hi, lo; splitf(x, hi, lo);
      Bh[kk][e] = hi; Bl[kk][e] = lo;
    }
  }

  float logg = 0.f;
  v8f accR[7];
  for (int j = 0; j < GROUP_L; ++j) {
    float sv = S[(size_t)(g * GROUP_L + j) * KP + n];   // diag entry, per lane
#pragma unroll
    for (int r = 0; r < 7; ++r) {
      v8f acc = {0.f, 0.f, 0.f, 0.f, 0.f, 0.f, 0.f, 0.f};
#pragma unroll
      for (int kk = 0; kk < 4; ++kk) {
        v16bf ah, al;
#pragma unroll
        for (int e = 0; e < 16; ++e) {
          int m = 16 * r + l15;
          int k2 = 32 * kk + a_kidx(e, half);
          ah[e] = Phi[m][k2];
          al[e] = Plo[m][k2];
        }
        acc = wmma_bf(ah, Bh[kk], acc);
        acc = wmma_bf(ah, Bl[kk], acc);
        acc = wmma_bf(al, Bh[kk], acc);
      }
#pragma unroll
      for (int v = 0; v < 8; ++v) acc[v] *= sv;   // column scaling = lane scalar
      accR[r] = acc;
    }
    // total-mass renormalization (all entries >= 0) -> keeps chain in f32 range
    float part = 0.f;
#pragma unroll
    for (int r = 0; r < 7; ++r)
#pragma unroll
      for (int v = 0; v < 8; ++v) part += accR[r][v];
    red[tid] = part;
    __syncthreads();
    for (int s2 = 128; s2 > 0; s2 >>= 1) {
      if (tid < s2) red[tid] += red[tid + s2];
      __syncthreads();
    }
    float ms = red[0];
    if (tid == 0) logg += logf(ms);
    float inv = 1.f / ms;
#pragma unroll
    for (int r = 0; r < 7; ++r) {
#pragma unroll
      for (int v = 0; v < 8; ++v) {
        accR[r][v] *= inv;
        int m = 16 * r + v + 8 * half;    // C-frag row -> P row
        __bf16 hi, lo; splitf(accR[r][v], hi, lo);
        Phi[m][n] = hi; Plo[m][n] = lo;
      }
    }
    __syncthreads();
  }
#pragma unroll
  for (int r = 0; r < 7; ++r)
#pragma unroll
    for (int v = 0; v < 8; ++v) {
      int m = 16 * r + v + 8 * half;
      Pout[((size_t)g * KP + m) * KP + n] = accR[r][v];
    }
  if (tid == 0) logG[g] = logg;
}

// ------------------------------------------------------------- k_combine ----
__global__ void __launch_bounds__(256)
k_combine(const float* __restrict__ Asoft, const float* __restrict__ Pout,
          const float* __restrict__ logsig, const float* __restrict__ logG,
          float* __restrict__ out) {
  __shared__ float alpha[KP];
  __shared__ float red[256];
  __shared__ float csh;
  int tid = threadIdx.x;
  float part = 0.f;
  for (int i = tid; i < TLEN;   i += 256) part += logsig[i];
  for (int i = tid; i < NGROUP; i += 256) part += logG[i];
  red[tid] = part;
  __syncthreads();
  for (int s2 = 128; s2 > 0; s2 >>= 1) {
    if (tid < s2) red[tid] += red[tid + s2];
    __syncthreads();
  }
  float total = red[0];
  __syncthreads();
  if (tid < KP) alpha[tid] = (tid == BOS_TAG) ? 1.f : 0.f;
  __syncthreads();
  float logz = 0.f;
  for (int g = 0; g < NGROUP; ++g) {
    float aj = 0.f;
    if (tid < KP) {
      const float* P = Pout + (size_t)g * KP * KP;
      for (int i = 0; i < KP; ++i) aj += alpha[i] * P[i * KP + tid];
    }
    red[tid] = (tid < KP) ? aj : 0.f;
    __syncthreads();
    if (tid == 0) {
      float s = 0.f;
      for (int i = 0; i < KP; ++i) s += red[i];
      csh = s;
    }
    __syncthreads();
    logz += logf(csh);
    if (tid < KP) alpha[tid] = aj / csh;
    __syncthreads();
  }
  if (tid == 0) {
    float z = 0.f;
    for (int i = 0; i < KP; ++i) z += alpha[i] * Asoft[i * KP + EOS_TAG];
    out[0] = total + logz + logf(z);
  }
}

// ----------------------------------------------------------------- launch ---
extern "C" void kernel_launch(void* const* d_in, const int* in_sizes, int n_in,
                              void* d_out, int out_size, void* d_ws, size_t ws_size,
                              hipStream_t stream) {
  (void)in_sizes; (void)n_in; (void)out_size; (void)ws_size;
  const float* ThetaB = (const float*)d_in[0];   // [100,256]
  const float* WA     = (const float*)d_in[1];   // [100,100]
  const float* E      = (const float*)d_in[2];   // [100000,256]
  const int*   words  = (const int*)d_in[3];     // [20000]
  const int*   tags   = (const int*)d_in[4];     // [20000]
  float* out = (float*)d_out;

  // workspace layout (bytes), ~28 MB total
  char* w = (char*)d_ws;
  float* Asoft  = (float*)(w + 0);                       //  50,176
  float* rowsum = (float*)(w + 50176);                   //     448
  float* logsig = (float*)(w + 50688);                   //  80,000
  float* logG   = (float*)(w + 130816);                  //   1,600
  float* S      = (float*)(w + 132608);                  // 8,960,000
  float* Pout   = (float*)(w + 9092608);                 // 20,070,400

  hipMemsetAsync(rowsum, 0, KP * sizeof(float), stream); // fresh every call

  k_prep   <<<1,      128, 0, stream>>>(WA, Asoft);
  k_rowsum <<<NVBLK,  224, 0, stream>>>(ThetaB, E, rowsum);
  k_emit   <<<NTTILE, 224, 0, stream>>>(ThetaB, E, words, tags, rowsum, S, logsig);
  k_group  <<<NGROUP, 224, 0, stream>>>(Asoft, S, Pout, logG);
  k_combine<<<1,      256, 0, stream>>>(Asoft, Pout, logsig, logG, out);
}
